// LSTMCell_83262236000379
// MI455X (gfx1250) — matrix operand order repaired and checked by reading, not compiled, not run on hardware
//
#include <hip/hip_runtime.h>
#include <hip/hip_bf16.h>
#include <cstdint>

// Problem constants
#define B_DIM 4096
#define DIN   1024
#define DH    1024
#define K_DIM 2048   // DIN + DH
#define NSTEP (K_DIM / 32)   // 64 K-steps of 32

typedef __attribute__((ext_vector_type(16))) __bf16 v16bf;
typedef __attribute__((ext_vector_type(8)))  float  v8f;

struct Q2 { uint4 a, b; };   // 32 bytes == v16bf

__device__ __forceinline__ unsigned short f32_to_bf16(float f) {
    union { float f; uint32_t u; } v; v.f = f;
    uint32_t u = v.u;
    u += 0x7FFFu + ((u >> 16) & 1u);   // round-to-nearest-even
    return (unsigned short)(u >> 16);
}

// Fast gate activations: single v_exp_f32 + v_rcp_f32, correct saturation at +-inf.
__device__ __forceinline__ float fast_exp(float x) {
    return __builtin_amdgcn_exp2f(x * 1.4426950408889634f);
}
__device__ __forceinline__ float fast_sigmoid(float x) {
    return __builtin_amdgcn_rcpf(1.0f + fast_exp(-x));   // x->-inf: 0, x->+inf: 1
}
__device__ __forceinline__ float fast_tanh(float x) {
    return 1.0f - 2.0f * __builtin_amdgcn_rcpf(1.0f + fast_exp(2.0f * x)); // +-1 limits
}

// CDNA5 async global->LDS copy (ASYNCcnt-tracked), GVS addressing mode.
__device__ __forceinline__ void async_ld16(uint32_t lds_off, const void* sbase, int voff) {
    asm volatile("global_load_async_to_lds_b128 %0, %1, %2"
                 :: "v"(lds_off), "v"(voff), "s"(sbase) : "memory");
}
__device__ __forceinline__ void wait_async0() {
#if __has_builtin(__builtin_amdgcn_s_wait_asynccnt)
    __builtin_amdgcn_s_wait_asynccnt(0);
#else
    asm volatile("s_wait_asynccnt 0x0" ::: "memory");
#endif
}

// ---------------------------------------------------------------------------
// Pre-pass 1: XH[m][k] = concat(x, h) converted to bf16, k-contiguous.
// ---------------------------------------------------------------------------
__global__ void __launch_bounds__(256)
pack_xh(const float* __restrict__ x, const float* __restrict__ h,
        unsigned short* __restrict__ XH) {
    int idx = (blockIdx.x * 256 + threadIdx.x) * 4;       // element index
    int m = idx >> 11;                                    // / 2048
    int k = idx & 2047;
    const float* src = (k < DIN) ? (x + (size_t)m * DIN + k)
                                 : (h + (size_t)m * DH + (k - DIN));
    float4 v = *(const float4*)src;
    ushort4 o;
    o.x = f32_to_bf16(v.x); o.y = f32_to_bf16(v.y);
    o.z = f32_to_bf16(v.z); o.w = f32_to_bf16(v.w);
    *(ushort4*)(XH + idx) = o;
}

// ---------------------------------------------------------------------------
// Pre-pass 2: WT[g*1024 + j][k] = W_g[k][j] in bf16 (n-major, k-contiguous).
// Tiled 32x32 transpose through LDS. Gate order: f, i, c, o.
// ---------------------------------------------------------------------------
__global__ void __launch_bounds__(256)
pack_w(const float* __restrict__ Wf, const float* __restrict__ Wi,
       const float* __restrict__ Wc, const float* __restrict__ Wo,
       unsigned short* __restrict__ WT) {
    __shared__ unsigned short tile[32][33];
    int g = blockIdx.z;
    const float* W = (g == 0) ? Wf : (g == 1) ? Wi : (g == 2) ? Wc : Wo;
    int j0 = blockIdx.x * 32;     // output-feature column base in W
    int k0 = blockIdx.y * 32;     // input row base in W
    int tx = threadIdx.x & 31;
    int ty = threadIdx.x >> 5;    // 0..7
#pragma unroll
    for (int s = 0; s < 4; ++s) {
        int k = k0 + ty + s * 8;
        tile[ty + s * 8][tx] = f32_to_bf16(W[(size_t)k * DH + j0 + tx]);
    }
    __syncthreads();
#pragma unroll
    for (int s = 0; s < 4; ++s) {
        int j = j0 + ty + s * 8;
        WT[(size_t)(g * DH + j) * K_DIM + k0 + tx] = tile[tx][ty + s * 8];
    }
}

// ---------------------------------------------------------------------------
// Fused GEMM + LSTM epilogue, double-buffered async global->LDS pipeline with
// STATIC buffer parity (K-loop unrolled by 2, last two steps peeled: no
// conditionals or pointer selection in the steady-state loop).
// Block tile: 128 rows (M) x 32 columns per gate (all 4 gates).
// 8 waves; each wave: 2 M-frags x 4 gates => 8 wmma_f32_16x16x32_bf16 / K-step.
// ---------------------------------------------------------------------------
__global__ void __launch_bounds__(256)
lstm_gemm(const unsigned short* __restrict__ XH,
          const unsigned short* __restrict__ WT,
          const float* __restrict__ c_in,
          const float* __restrict__ b_f, const float* __restrict__ b_i,
          const float* __restrict__ b_c, const float* __restrict__ b_o,
          float* __restrict__ out_h, float* __restrict__ out_c) {
    __shared__ __align__(16) unsigned short Atile[2][128 * 32];
    __shared__ __align__(16) unsigned short Btile[2][128 * 32];

    const int tid  = threadIdx.x;
    const int n0   = blockIdx.x * 32;   // gate-column base (0..1023)
    const int m0   = blockIdx.y * 128;  // batch-row base
    const int wave = tid >> 5;
    const int lane = tid & 31;
    const int wm   = wave & 3;          // M split (4)
    const int sub  = (wave >> 2) * 16;  // N split within 32 cols (2)
    const int half = lane >> 4;
    const int l15  = lane & 15;

    // ---- per-thread fill chunk coordinates (two 16B chunks per tile)
    int gOffA[2], gOffB[2];             // global byte offsets (k excluded)
    uint32_t ldsA[2][2], ldsB[2][2];    // [chunk][buffer] LDS byte addresses
#pragma unroll
    for (int cc = 0; cc < 2; ++cc) {
        int c   = tid + cc * 256;
        int t   = c >> 2;               // tile row 0..127
        int off = (c & 3) * 8;          // element offset within 32-wide k
        gOffA[cc] = ((m0 + t) * K_DIM + off) * 2;
        int g     = t >> 5;
        int wtrow = g * DH + n0 + (t & 31);
        gOffB[cc] = (wtrow * K_DIM + off) * 2;
#pragma unroll
        for (int p = 0; p < 2; ++p) {
            ldsA[cc][p] = (uint32_t)(uintptr_t)&Atile[p][t * 32 + off];
            ldsB[cc][p] = (uint32_t)(uintptr_t)&Btile[p][t * 32 + off];
        }
    }

    v8f acc[2][4];
#pragma unroll
    for (int mi = 0; mi < 2; ++mi)
#pragma unroll
        for (int g = 0; g < 4; ++g)
            acc[mi][g] = (v8f){0.f, 0.f, 0.f, 0.f, 0.f, 0.f, 0.f, 0.f};

    // Async-fill buffer p with K-step `step` (32 k-elements = 64 bytes each).
    auto fill = [&](int p, int step) {
        int kb = step * 64;
#pragma unroll
        for (int cc = 0; cc < 2; ++cc) {
            async_ld16(ldsA[cc][p], XH, gOffA[cc] + kb);
            async_ld16(ldsB[cc][p], WT, gOffB[cc] + kb);
        }
    };

    // Consume buffer p: build fragments per ISA VGPR layouts, 8 WMMAs.
    auto compute = [&](int p) {
        const unsigned short* Ab = Atile[p];
        const unsigned short* Bb = Btile[p];
        v16bf afrag[2], bfrag[4];
#pragma unroll
        for (int mi = 0; mi < 2; ++mi) {
            int row = wm * 32 + mi * 16 + l15;
            // A 16x32: lanes 0-15 k={0..7,16..23}, lanes 16-31 k={8..15,24..31}
            const uint4* pa = (const uint4*)(Ab + row * 32 + half * 8);
            Q2 q; q.a = pa[0]; q.b = pa[2];         // +32B == k+16 elements
            afrag[mi] = __builtin_bit_cast(v16bf, q);
        }
#pragma unroll
        for (int g = 0; g < 4; ++g) {
            int row = g * 32 + sub + l15;           // B-tile row == output col
            // B 32x16: lanes 0-15 k={0..7,16..23}, lanes 16-31 k={8..15,24..31}
            const uint4* pb = (const uint4*)(Bb + row * 32 + half * 8);
            Q2 q; q.a = pb[0]; q.b = pb[2];
            bfrag[g] = __builtin_bit_cast(v16bf, q);
        }
#pragma unroll
        for (int mi = 0; mi < 2; ++mi)
#pragma unroll
            for (int g = 0; g < 4; ++g)
                acc[mi][g] = __builtin_amdgcn_wmma_f32_16x16x32_bf16(
                    false, afrag[mi], false, bfrag[g],
                    (short)0, acc[mi][g], false, false);
    };

    // ---- pipeline: prologue fill, branch-free steady state, peeled tail
    fill(0, 0);
    wait_async0();
    __syncthreads();

    for (int s = 0; s < NSTEP - 2; s += 2) {
        fill(1, s + 1);         // prefetch next step into buf 1
        compute(0);             // consume buf 0 (step s)
        wait_async0();
        __syncthreads();
        fill(0, s + 2);         // prefetch into buf 0
        compute(1);             // consume buf 1 (step s+1)
        wait_async0();
        __syncthreads();
    }
    fill(1, NSTEP - 1);         // last prefetch
    compute(0);                 // step NSTEP-2
    wait_async0();
    __syncthreads();
    compute(1);                 // step NSTEP-1

    // ---- fused LSTM epilogue (all 4 gates for the same column in-register)
    const int j  = n0 + sub + l15;                 // output column 0..1023
    const float bf = b_f[j], bi = b_i[j], bc = b_c[j], bo = b_o[j];
#pragma unroll
    for (int mi = 0; mi < 2; ++mi) {
#pragma unroll
        for (int r = 0; r < 8; ++r) {
            int m = m0 + wm * 32 + mi * 16 + half * 8 + r;  // C layout: M=r+8*half
            float f  = fast_sigmoid(acc[mi][0][r] + bf);
            float i  = fast_sigmoid(acc[mi][1][r] + bi);
            float cs = fast_tanh   (acc[mi][2][r] + bc);
            float o  = fast_sigmoid(acc[mi][3][r] + bo);
            float cp = c_in[(size_t)m * DH + j];
            float ct = f * cp + i * cs;
            float ht = o * fast_tanh(ct);
            out_h[(size_t)m * DH + j] = ht;
            out_c[(size_t)m * DH + j] = ct;
        }
    }
}

// ---------------------------------------------------------------------------
extern "C" void kernel_launch(void* const* d_in, const int* in_sizes, int n_in,
                              void* d_out, int out_size, void* d_ws, size_t ws_size,
                              hipStream_t stream) {
    (void)in_sizes; (void)n_in; (void)out_size; (void)ws_size;
    const float* x  = (const float*)d_in[0];
    const float* h  = (const float*)d_in[1];
    const float* c  = (const float*)d_in[2];
    const float* Wi = (const float*)d_in[3];
    const float* Wf = (const float*)d_in[4];
    const float* Wc = (const float*)d_in[5];
    const float* Wo = (const float*)d_in[6];
    const float* bi = (const float*)d_in[7];
    const float* bf = (const float*)d_in[8];
    const float* bc = (const float*)d_in[9];
    const float* bo = (const float*)d_in[10];

    unsigned short* XH = (unsigned short*)d_ws;                 // 4096*2048 bf16
    unsigned short* WT = XH + (size_t)B_DIM * K_DIM;            // 4096*2048 bf16

    float* out_h = (float*)d_out;
    float* out_c = out_h + (size_t)B_DIM * DH;

    pack_xh<<<(B_DIM * K_DIM / 4) / 256, 256, 0, stream>>>(x, h, XH);
    pack_w<<<dim3(DH / 32, K_DIM / 32, 4), 256, 0, stream>>>(Wf, Wi, Wc, Wo, WT);
    lstm_gemm<<<dim3(DH / 32, B_DIM / 128), 256, 0, stream>>>(
        XH, WT, c, bf, bi, bc, bo, out_h, out_c);
}
